// LinearTransformer_86526411145271
// MI455X (gfx1250) — matrix operand order repaired and checked by reading, not compiled, not run on hardware
//
#include <hip/hip_runtime.h>
#include <hip/hip_bf16.h>

// ---------------- types for CDNA5 WMMA ----------------
typedef __attribute__((ext_vector_type(16))) __bf16 v16bf;
typedef __attribute__((ext_vector_type(8)))  float  v8f;
typedef __attribute__((ext_vector_type(8)))  unsigned int v8u;

#define EPSF 1e-6f
#define EPSL 1e-5f

// pack two fp32 into one dword of 2 bf16 (truncation): lo -> bits[15:0]. One v_perm_b32.
__device__ __forceinline__ unsigned pack2(float lo, float hi) {
    return __builtin_amdgcn_perm(__builtin_bit_cast(unsigned, hi),
                                 __builtin_bit_cast(unsigned, lo), 0x07060302u);
}
__device__ __forceinline__ float reluq(float x) { return fmaxf(x, 0.f) + EPSF; }

__device__ __forceinline__ v8f wmma_bf16(v8u a, v8u b, v8f c) {
    return __builtin_amdgcn_wmma_f32_16x16x32_bf16(
        false, __builtin_bit_cast(v16bf, a),
        false, __builtin_bit_cast(v16bf, b),
        (short)0, c, false, false);
}

// =======================================================================
// input projection: x[n,e] = obs[n,:]@Win + bin[e] + pos_enc(n%1024, e)
// =======================================================================
__global__ void input_proj_k(const float* __restrict__ obs,
                             const float* __restrict__ Win,
                             const float* __restrict__ bin,
                             float* __restrict__ X) {
    int tid = blockIdx.x * 256 + threadIdx.x;            // 16384*512
    int n = tid >> 9, e = tid & 511;
    int s = n & 1023;
    const float* o = obs + (size_t)n * 3;
    float v = bin[e] + o[0] * Win[e] + o[1] * Win[512 + e] + o[2] * Win[1024 + e];
    float freq = __expf(-(float)(e & ~1) * (9.210340371976184f / 512.0f));
    float ang = (float)s * freq;
    v += (e & 1) ? __cosf(ang) : __sinf(ang);
    X[tid] = v;
}

// =======================================================================
// weight pre-pack: W[K,N] fp32 -> bf16 fragments in wave32 B layout
// P index = ((kt*TN + tn)*32 + lane)*8 + v2 ; pair k = kt*32+(lane>>4)*16+2*v2
// =======================================================================
template<int N, int K>
__global__ void packw_k(const float* __restrict__ W, unsigned* __restrict__ P) {
    constexpr int TN = N / 16;
    int t = blockIdx.x * 256 + threadIdx.x;              // K*N/2 dwords
    int v2 = t & 7, lane = (t >> 3) & 31, r = t >> 8;    // r = kt*TN + tn
    int tn = r % TN, kt = r / TN;
    int kb = kt * 32 + ((lane >> 4) << 4) + (v2 << 1);
    int col = tn * 16 + (lane & 15);
    P[t] = pack2(W[(size_t)kb * N + col], W[(size_t)(kb + 1) * N + col]);
}

// =======================================================================
// GEMM: C[M,N] = A[M,K] @ W + bias (+res).
// Block = 8 waves x 64-row A macro-tile (4 row-tiles), staged via LDS
// (double-buffered, 1 barrier/K-step). Each wave: one 16-col strip,
// 4 accumulators; one pre-packed B fragment (2xb128) feeds 4 WMMAs.
// grid = (M/64) * (N/16/8) blocks of 256 threads.
// =======================================================================
template<int N, int K>
__global__ void gemm_bf16_k(const float* __restrict__ A,
                            const unsigned* __restrict__ Wpk,
                            const float* __restrict__ bias,
                            const float* __restrict__ res,
                            float* __restrict__ C) {
    constexpr int TN = N / 16;
    __shared__ __align__(32) unsigned sA[2][1024];
    int wid  = threadIdx.x >> 5;
    int lane = threadIdx.x & 31;
    int bm = blockIdx.x / (TN / 8);
    int tn = (blockIdx.x % (TN / 8)) * 8 + wid;

    // A staging role: thread t packs dword (slane, sv) for 4 row-tiles
    int slane = threadIdx.x >> 3, sv = threadIdx.x & 7;
    int sk = ((slane >> 4) << 4) + (sv << 1);
    const float* Abase = A + (size_t)(bm * 64 + (slane & 15)) * K + sk;

    v8f acc0 = {0.f,0.f,0.f,0.f,0.f,0.f,0.f,0.f};
    v8f acc1 = acc0, acc2 = acc0, acc3 = acc0;
    const unsigned* bp = Wpk + ((size_t)tn * 32 + lane) * 8;
#pragma unroll 1
    for (int kt = 0; kt < K / 32; ++kt) {
        int buf = kt & 1;
#pragma unroll
        for (int rt = 0; rt < 4; ++rt) {
            float2 f = *(const float2*)(Abase + (size_t)(rt * 16) * K + kt * 32);
            sA[buf][rt * 256 + threadIdx.x] = pack2(f.x, f.y);
        }
        __syncthreads();
        v8u bu = *(const v8u*)(bp + (size_t)kt * TN * 256);      // 2x global_load_b128
        __builtin_prefetch(bp + (size_t)(kt + 1) * TN * 256, 0, 1);
        v8u a0 = *(const v8u*)(&sA[buf][lane * 8]);              // 2x ds_load_b128 each
        v8u a1 = *(const v8u*)(&sA[buf][256 + lane * 8]);
        v8u a2 = *(const v8u*)(&sA[buf][512 + lane * 8]);
        v8u a3 = *(const v8u*)(&sA[buf][768 + lane * 8]);
        acc0 = wmma_bf16(a0, bu, acc0);
        acc1 = wmma_bf16(a1, bu, acc1);
        acc2 = wmma_bf16(a2, bu, acc2);
        acc3 = wmma_bf16(a3, bu, acc3);
    }
    int col  = tn * 16 + (lane & 15);
    int row0 = bm * 64 + ((lane >> 4) << 3);
    float bv = bias[col];
    float* cp = C + (size_t)row0 * N + col;
    v8f* accs[4] = {&acc0, &acc1, &acc2, &acc3};
    if (res) {
        const float* rp = res + (size_t)row0 * N + col;
#pragma unroll
        for (int rt = 0; rt < 4; ++rt)
#pragma unroll
            for (int v2 = 0; v2 < 8; ++v2)
                cp[(rt * 16 + v2) * N] = (*accs[rt])[v2] + bv + rp[(rt * 16 + v2) * N];
    } else {
#pragma unroll
        for (int rt = 0; rt < 4; ++rt)
#pragma unroll
            for (int v2 = 0; v2 < 8; ++v2)
                cp[(rt * 16 + v2) * N] = (*accs[rt])[v2] + bv;
    }
}

// =======================================================================
// attention: kv[bh,d,e] = sum_n relu(k[n,d]) * v[n,e]   (64x64 per bh)
// qkv layout: [B*S,1536], cols [0:512]=q [512:1024]=k [1024:1536]=v
// =======================================================================
__global__ void attn_kv_k(const float* __restrict__ qkv, float* __restrict__ kvb) {
    int wave = blockIdx.x * 8 + (threadIdx.x >> 5);      // 2048 waves
    int lane = threadIdx.x & 31;
    int bh = wave >> 4, t = wave & 15;
    int td = t >> 2, te = t & 3;
    int b = bh >> 3, h = bh & 7;
    const float* kbase = qkv + (size_t)b * 1572864 + 512 + h * 64;
    const float* vbase = qkv + (size_t)b * 1572864 + 1024 + h * 64;
    int ah = lane >> 4;
    int dcol = td * 16 + (lane & 15);
    int ecol = te * 16 + (lane & 15);
    int bk0 = (lane >> 4) << 4;

    v8f acc = {0.f,0.f,0.f,0.f,0.f,0.f,0.f,0.f};
    for (int n0 = 0; n0 < 1024; n0 += 32) {
        v8u au, bu;
#pragma unroll
        for (int v2 = 0; v2 < 8; ++v2) {
            int ka = n0 + (ah << 4) + (v2 << 1);
            au[v2] = pack2(reluq(kbase[(size_t)ka * 1536 + dcol]),
                           reluq(kbase[(size_t)(ka + 1) * 1536 + dcol]));
            int kb = n0 + bk0 + (v2 << 1);
            bu[v2] = pack2(vbase[(size_t)kb * 1536 + ecol],
                           vbase[(size_t)(kb + 1) * 1536 + ecol]);
        }
        acc = wmma_bf16(au, bu, acc);
    }
#pragma unroll
    for (int v2 = 0; v2 < 8; ++v2) {
        int row = td * 16 + v2 + ((lane >> 4) << 3);
        kvb[(size_t)bh * 4096 + row * 64 + ecol] = acc[v2];
    }
}

// ksum[bh,d] = sum_n relu(k[n,d])
__global__ void ksum_k(const float* __restrict__ qkv, float* __restrict__ ksum) {
    int tid = blockIdx.x * 256 + threadIdx.x;            // 8192
    int bh = tid >> 6, d = tid & 63;
    int b = bh >> 3, h = bh & 7;
    const float* kb = qkv + (size_t)b * 1572864 + 512 + h * 64 + d;
    float s = 0.f;
    for (int n = 0; n < 1024; ++n) s += reluq(kb[(size_t)n * 1536]);
    ksum[tid] = s;
}

// norm[bh,n] = relu(q[n,:]) . ksum[bh,:] + eps
__global__ void norm_k(const float* __restrict__ qkv, const float* __restrict__ ksum,
                       float* __restrict__ nrm) {
    int tid = blockIdx.x * 256 + threadIdx.x;            // 131072
    int bh = tid >> 10, n = tid & 1023;
    int b = bh >> 3, h = bh & 7;
    const float* qb = qkv + (size_t)(b * 1024 + n) * 1536 + h * 64;
    const float* ks = ksum + bh * 64;
    float s = 0.f;
#pragma unroll 8
    for (int d = 0; d < 64; ++d) s += reluq(qb[d]) * ks[d];
    nrm[tid] = s + EPSF;
}

// out[n, h*64+e] = (relu(q[n,:]) @ kv[bh]) / norm[bh,n]
__global__ void attn_out_k(const float* __restrict__ qkv, const float* __restrict__ kvb,
                           const float* __restrict__ nrm, float* __restrict__ att) {
    int wave = blockIdx.x * 8 + (threadIdx.x >> 5);      // 32768 waves
    int lane = threadIdx.x & 31;
    int bh = wave >> 8, t = wave & 255;
    int tmm = t >> 2, te = t & 3;
    int b = bh >> 3, h = bh & 7;
    const float* qbase = qkv + (size_t)b * 1572864 + h * 64;
    int ah = lane >> 4;
    int nrow = tmm * 16 + (lane & 15);
    int ecol = te * 16 + (lane & 15);
    int bk0 = (lane >> 4) << 4;

    v8f acc = {0.f,0.f,0.f,0.f,0.f,0.f,0.f,0.f};
#pragma unroll
    for (int k0 = 0; k0 < 64; k0 += 32) {
        v8u au, bu;
#pragma unroll
        for (int v2 = 0; v2 < 8; ++v2) {
            int ka = k0 + (ah << 4) + (v2 << 1);
            float2 f = *(const float2*)(qbase + (size_t)nrow * 1536 + ka);
            au[v2] = pack2(reluq(f.x), reluq(f.y));
            int kb = k0 + bk0 + (v2 << 1);
            bu[v2] = pack2(kvb[(size_t)bh * 4096 + kb * 64 + ecol],
                           kvb[(size_t)bh * 4096 + (kb + 1) * 64 + ecol]);
        }
        acc = wmma_bf16(au, bu, acc);
    }
#pragma unroll
    for (int v2 = 0; v2 < 8; ++v2) {
        int n = tmm * 16 + v2 + ((lane >> 4) << 3);
        float r = acc[v2] / nrm[bh * 1024 + n];
        att[(size_t)(b * 1024 + n) * 512 + h * 64 + ecol] = r;
    }
}

// =======================================================================
// layernorm over 512 cols
// =======================================================================
__global__ void ln_k(const float* __restrict__ src, float* __restrict__ dst,
                     const float* __restrict__ sc, const float* __restrict__ of) {
    __shared__ float s1[256], s2[256];
    int row = blockIdx.x, t = threadIdx.x;
    const float* r = src + (size_t)row * 512;
    float a = r[t], b = r[t + 256];
    s1[t] = a + b; s2[t] = a * a + b * b;
    __syncthreads();
    for (int s = 128; s > 0; s >>= 1) {
        if (t < s) { s1[t] += s1[t + s]; s2[t] += s2[t + s]; }
        __syncthreads();
    }
    float mu = s1[0] * (1.f / 512.f);
    float inv = rsqrtf(s2[0] * (1.f / 512.f) - mu * mu + EPSL);
    dst[(size_t)row * 512 + t]       = (a - mu) * inv * sc[t] + of[t];
    dst[(size_t)row * 512 + t + 256] = (b - mu) * inv * sc[t + 256] + of[t + 256];
}

__global__ void mean_k(const float* __restrict__ x, float* __restrict__ emb) {
    int tid = blockIdx.x * 256 + threadIdx.x;            // 8192
    int b = tid >> 9, e = tid & 511;
    const float* p = x + (size_t)b * 524288 + e;
    float s = 0.f;
    for (int n = 0; n < 1024; ++n) s += p[(size_t)n * 512];
    emb[tid] = s * (1.f / 1024.f);
}

__global__ void head_k(const float* __restrict__ emb, const float* __restrict__ Wh,
                       const float* __restrict__ bh, float* __restrict__ out) {
    int tid = blockIdx.x * 256 + threadIdx.x;            // 4096
    int b = tid >> 8, o = tid & 255;
    const float* e = emb + b * 512;
    float s = bh[o];
    for (int k = 0; k < 512; ++k) s += e[k] * Wh[(size_t)k * 256 + o];
    out[tid] = s;
}

// =======================================================================
extern "C" void kernel_launch(void* const* d_in, const int* in_sizes, int n_in,
                              void* d_out, int out_size, void* d_ws, size_t ws_size,
                              hipStream_t stream) {
    (void)in_sizes; (void)n_in; (void)out_size; (void)ws_size;
    const float* obs = (const float*)d_in[0];
    const float* Win = (const float*)d_in[1];
    const float* bin = (const float*)d_in[2];

    float* ws   = (float*)d_ws;
    float* X    = ws;                               // 16384*512
    float* QKV  = X    + (size_t)16384 * 512;       // 16384*1536
    float* ATT  = QKV  + (size_t)16384 * 1536;      // 16384*512
    float* KV   = ATT  + (size_t)16384 * 512;       // 128*4096
    float* KSUM = KV   + (size_t)128 * 4096;        // 8192
    float* NRM  = KSUM + 8192;                      // 131072
    float* EMB  = NRM  + 131072;                    // 8192
    unsigned* PKW = (unsigned*)(EMB + 8192);        // 393216 dwords (packed weights)

    input_proj_k<<<32768, 256, 0, stream>>>(obs, Win, bin, X);

    for (int i = 0; i < 3; ++i) {
        const float* Wq  = (const float*)d_in[3 + i * 12 + 0];
        const float* bq  = (const float*)d_in[3 + i * 12 + 1];
        const float* Wp  = (const float*)d_in[3 + i * 12 + 2];
        const float* bp  = (const float*)d_in[3 + i * 12 + 3];
        const float* lAs = (const float*)d_in[3 + i * 12 + 4];
        const float* lAo = (const float*)d_in[3 + i * 12 + 5];
        // mlp1 (indices +6,+7) is dead code in the reference: skipped.
        const float* W2  = (const float*)d_in[3 + i * 12 + 8];
        const float* b2  = (const float*)d_in[3 + i * 12 + 9];
        const float* lMs = (const float*)d_in[3 + i * 12 + 10];
        const float* lMo = (const float*)d_in[3 + i * 12 + 11];

        // qkv = x @ Wq + bq                      [16384,1536]
        packw_k<1536, 512><<<1536, 256, 0, stream>>>(Wq, PKW);
        gemm_bf16_k<1536, 512><<<3072, 256, 0, stream>>>(X, PKW, bq, nullptr, QKV);
        // linear attention
        ksum_k    <<<32,   256, 0, stream>>>(QKV, KSUM);
        attn_kv_k <<<256,  256, 0, stream>>>(QKV, KV);
        norm_k    <<<512,  256, 0, stream>>>(QKV, KSUM, NRM);
        attn_out_k<<<4096, 256, 0, stream>>>(QKV, KV, NRM, ATT);
        // x = x + attn @ Wp + bp ; LN
        packw_k<512, 512><<<512, 256, 0, stream>>>(Wp, PKW);
        gemm_bf16_k<512, 512><<<1024, 256, 0, stream>>>(ATT, PKW, bp, X, X);
        ln_k<<<16384, 256, 0, stream>>>(X, X, lAs, lAo);
        // tmp = x + x @ W2 + b2 (into ATT, avoids in-place RAW) ; LN back into X
        packw_k<512, 512><<<512, 256, 0, stream>>>(W2, PKW);
        gemm_bf16_k<512, 512><<<1024, 256, 0, stream>>>(X, PKW, b2, X, ATT);
        ln_k<<<16384, 256, 0, stream>>>(ATT, X, lMs, lMo);
    }

    mean_k<<<32, 256, 0, stream>>>(X, EMB);
    head_k<<<16, 256, 0, stream>>>(EMB, (const float*)d_in[39], (const float*)d_in[40],
                                   (float*)d_out);
}
